// VisionRetentionChunk_9182640078864
// MI455X (gfx1250) — matrix operand order repaired and checked by reading, not compiled, not run on hardware
//
#include <hip/hip_runtime.h>
#include <hip/hip_bf16.h>
#include <math.h>

// ---------------------------------------------------------------------------
// VisionRetentionChunk on MI455X (gfx1250, wave32, WMMA 16x16x32 f16).
// Memory-bound (~45 GFLOP vs ~0.5GB @ 23.3TB/s): keep intermediates fp16,
// run every matmul through v_wmma_f32_16x16x32_f16 with fp32 accumulation,
// fuse bias/scale/mask into the WMMA C operand.  V / vv tiles are staged
// into LDS with gfx1250 async copies (GLOBAL_LOAD_ASYNC_TO_LDS_B128,
// ASYNCcnt) overlapped with the QK^T WMMA + softmax.
// ---------------------------------------------------------------------------

#define Bx 16
#define Hx 64
#define Wx 64
#define Cx 256
#define NHx 8
#define KDx 32
#define NROW (Bx*Hx*Wx)            // 65536 GEMM rows
#define SCALE_F 0.17677669529663687f

typedef _Float16 h16 __attribute__((ext_vector_type(16)));
typedef _Float16 h8  __attribute__((ext_vector_type(8)));
typedef float    f8  __attribute__((ext_vector_type(8)));
typedef float    f4  __attribute__((ext_vector_type(4)));

union H16 { h16 v; h8 h2[2]; _Float16 e[16]; };
union F8  { f8 v; float e[8]; };

// ---- gfx1250 async global->LDS copy helpers -------------------------------
typedef __attribute__((address_space(3))) const _Float16 lds_cf16;

__device__ __forceinline__ unsigned lds_offset(const _Float16* p) {
    // generic -> LDS addrspace cast yields the 32-bit LDS byte offset
    return (unsigned)(uintptr_t)(lds_cf16*)p;
}
__device__ __forceinline__ void async_ld_b128(unsigned ldsoff, const _Float16* g) {
    asm volatile("global_load_async_to_lds_b128 %0, %1, off"
                 :: "v"(ldsoff), "v"((unsigned long long)(uintptr_t)g)
                 : "memory");
}
__device__ __forceinline__ void wait_async0() {
    asm volatile("s_wait_asynccnt 0x0" ::: "memory");
}

// ---------------------------------------------------------------------------
// Pack 256x256 fp32 weights W[k][n] into f16 B-operand layout:
// pk[(k>>4)*4096 + n*16 + (k&15)]  -> a lane's 16 K-values are contiguous.
// ---------------------------------------------------------------------------
__global__ void pack_w_kernel(const float* __restrict__ W0, const float* __restrict__ W1,
                              const float* __restrict__ W2, const float* __restrict__ W3,
                              _Float16* __restrict__ pk)
{
    int idx  = blockIdx.x * 256 + threadIdx.x;      // [0, 4*65536)
    int wsel = idx >> 16;
    int rem  = idx & 65535;
    int k    = rem >> 8;
    int n    = rem & 255;
    const float* W = (wsel == 0) ? W0 : (wsel == 1) ? W1 : (wsel == 2) ? W2 : W3;
    pk[(size_t)wsel * 65536 + (size_t)(k >> 4) * 4096 + n * 16 + (k & 15)] =
        (_Float16)W[k * 256 + n];
}

// ---------------------------------------------------------------------------
// Fused Q/K/V projection GEMM: (65536x256)x(256x256)x3, bias in C operand,
// K scaled by 1/sqrt(KD).  Block = 4 waves; wave = 16 rows x 64 cols.
// A fragments (8 k-steps) loaded once, reused across the 3 projections.
// ---------------------------------------------------------------------------
__global__ __launch_bounds__(128)
void proj_qkv_kernel(const float* __restrict__ x,
                     const _Float16* __restrict__ Wq_pk, const float* __restrict__ bq,
                     const _Float16* __restrict__ Wk_pk, const float* __restrict__ bk,
                     const _Float16* __restrict__ Wv_pk, const float* __restrict__ bv,
                     _Float16* __restrict__ Qh, _Float16* __restrict__ Kh,
                     _Float16* __restrict__ Vh)
{
    const int lane = threadIdx.x & 31;
    const int wave = threadIdx.x >> 5;
    const int mi   = lane & 15;
    const int khf  = lane >> 4;
    const int rowBase = blockIdx.x * 64 + wave * 16;
    const int colBase = blockIdx.y * 64;

    // A fragments: 16x32 f16 per k-step; element i <-> k = 16*khf+i
    H16 a[8];
    const float* arow = x + (size_t)(rowBase + mi) * Cx;
    #pragma unroll
    for (int s = 0; s < 8; ++s) {
        const float* p0 = arow + 32 * s + 16 * khf;
        f4 lo0 = *(const f4*)(p0);
        f4 lo1 = *(const f4*)(p0 + 4);
        f4 hi0 = *(const f4*)(p0 + 8);
        f4 hi1 = *(const f4*)(p0 + 12);
        #pragma unroll
        for (int i = 0; i < 4; ++i) {
            a[s].e[i]      = (_Float16)lo0[i];
            a[s].e[4 + i]  = (_Float16)lo1[i];
            a[s].e[8 + i]  = (_Float16)hi0[i];
            a[s].e[12 + i] = (_Float16)hi1[i];
        }
    }

    const _Float16* Wpk[3] = { Wq_pk, Wk_pk, Wv_pk };
    const float*    bia[3] = { bq, bk, bv };
    _Float16*       Out[3] = { Qh, Kh, Vh };
    const float     scl[3] = { 1.0f, SCALE_F, 1.0f };

    #pragma unroll
    for (int p = 0; p < 3; ++p) {
        F8 acc[4];
        #pragma unroll
        for (int j = 0; j < 4; ++j) {
            float bval = bia[p][colBase + 16 * j + mi];
            #pragma unroll
            for (int r = 0; r < 8; ++r) acc[j].e[r] = bval;
        }
        #pragma unroll
        for (int s = 0; s < 8; ++s) {
            #pragma unroll
            for (int j = 0; j < 4; ++j) {
                const _Float16* bp = Wpk[p] + (size_t)(2 * s + khf) * 4096
                                            + (size_t)(colBase + 16 * j + mi) * 16;
                H16 bf;
                bf.h2[0] = *(const h8*)(bp);
                bf.h2[1] = *(const h8*)(bp + 8);
                acc[j].v = __builtin_amdgcn_wmma_f32_16x16x32_f16(
                    false, a[s].v, false, bf.v, (short)0, acc[j].v, false, false);
            }
        }
        #pragma unroll
        for (int j = 0; j < 4; ++j) {
            int col = colBase + 16 * j + mi;
            #pragma unroll
            for (int r = 0; r < 8; ++r) {
                int row = rowBase + r + 8 * khf;
                Out[p][(size_t)row * Cx + col] = (_Float16)(acc[j].e[r] * scl[p]);
            }
        }
    }
}

// ---------------------------------------------------------------------------
// In-place rotary (theta_shift) on Q and K. One thread per even/odd pair.
// ---------------------------------------------------------------------------
__global__ void theta_kernel(_Float16* __restrict__ Qh, _Float16* __restrict__ Kh,
                             const float* __restrict__ sinp, const float* __restrict__ cosp)
{
    size_t e = ((size_t)blockIdx.x * 256 + threadIdx.x) * 2;   // element index
    int c = (int)(e & 255);
    int d = c & 31;
    size_t row = e >> 8;                 // (b*64+h)*64+w
    int w = (int)(row & 63);
    int h = (int)((row >> 6) & 63);
    size_t si = ((size_t)(h * 64 + w)) * 32 + d;
    float s0 = sinp[si], s1 = sinp[si + 1];
    float c0 = cosp[si], c1 = cosp[si + 1];

    float q0 = (float)Qh[e], q1 = (float)Qh[e + 1];
    Qh[e]     = (_Float16)(q0 * c0 - q1 * s0);
    Qh[e + 1] = (_Float16)(q1 * c1 + q0 * s1);

    float k0 = (float)Kh[e], k1 = (float)Kh[e + 1];
    Kh[e]     = (_Float16)(k0 * c0 - k1 * s0);
    Kh[e + 1] = (_Float16)(k1 * c1 + k0 * s1);
}

// ---------------------------------------------------------------------------
// Attention along W: one block per (b, h, head). 4 waves x 16 query rows.
// V tile (64x32 f16) async-copied to LDS at kernel entry, overlapped with
// S = QK^T + mask (mask in WMMA C), shfl softmax, then O = P x V.
// Output stored to vvT[b][w][n][h][d] for the H-axis pass.
// ---------------------------------------------------------------------------
__global__ __launch_bounds__(128)
void attn_w_kernel(const _Float16* __restrict__ Qh, const _Float16* __restrict__ Kh,
                   const _Float16* __restrict__ Vh, const float* __restrict__ mask_w,
                   _Float16* __restrict__ vvT)
{
    __shared__ _Float16 lds[4][16][64];
    __shared__ _Float16 vtile[64][32];            // [w''][d]
    const int lane = threadIdx.x & 31;
    const int wave = threadIdx.x >> 5;
    const int mi   = lane & 15;
    const int khf  = lane >> 4;
    const int bid  = blockIdx.x;                 // (b*64+h)*8+n
    const int n = bid & 7, h = (bid >> 3) & 63, b = bid >> 9;
    const int wBase = wave * 16;
    const size_t plane = ((size_t)(b * 64 + h)) * 64 * 256 + (size_t)n * 32;

    // kick off async V-tile copy (2 x b128 per thread, ASYNCcnt-tracked)
    {
        int t  = threadIdx.x;
        int wv = t >> 1;
        int d0 = (t & 1) * 16;
        const _Float16* g = Vh + plane + (size_t)wv * 256 + d0;
        unsigned lo = lds_offset(&vtile[wv][d0]);
        async_ld_b128(lo, g);
        async_ld_b128(lo + 16, g + 8);
    }

    H16 a;
    {
        const _Float16* p = Qh + plane + (size_t)(wBase + mi) * 256 + 16 * khf;
        a.h2[0] = *(const h8*)(p);
        a.h2[1] = *(const h8*)(p + 8);
    }
    F8 acc[4];
    #pragma unroll
    for (int j = 0; j < 4; ++j) {
        int v = 16 * j + mi;
        #pragma unroll
        for (int r = 0; r < 8; ++r)
            acc[j].e[r] = mask_w[((size_t)n * 64 + (wBase + r + 8 * khf)) * 64 + v];
        H16 bf;
        const _Float16* kp = Kh + plane + (size_t)v * 256 + 16 * khf;
        bf.h2[0] = *(const h8*)(kp);
        bf.h2[1] = *(const h8*)(kp + 8);
        acc[j].v = __builtin_amdgcn_wmma_f32_16x16x32_f16(
            false, a.v, false, bf.v, (short)0, acc[j].v, false, false);
    }
    #pragma unroll
    for (int r = 0; r < 8; ++r) {
        float mx = fmaxf(fmaxf(acc[0].e[r], acc[1].e[r]), fmaxf(acc[2].e[r], acc[3].e[r]));
        #pragma unroll
        for (int off = 1; off < 16; off <<= 1) mx = fmaxf(mx, __shfl_xor(mx, off, 32));
        float sum = 0.f;
        #pragma unroll
        for (int j = 0; j < 4; ++j) { acc[j].e[r] = __expf(acc[j].e[r] - mx); sum += acc[j].e[r]; }
        #pragma unroll
        for (int off = 1; off < 16; off <<= 1) sum += __shfl_xor(sum, off, 32);
        float inv = 1.f / sum;
        #pragma unroll
        for (int j = 0; j < 4; ++j) acc[j].e[r] *= inv;
    }
    #pragma unroll
    for (int j = 0; j < 4; ++j)
        #pragma unroll
        for (int r = 0; r < 8; ++r)
            lds[wave][r + 8 * khf][16 * j + mi] = (_Float16)acc[j].e[r];

    wait_async0();          // V tile resident in LDS
    __syncthreads();        // visible block-wide (and P strips committed)

    F8 o[2];
    #pragma unroll
    for (int j2 = 0; j2 < 2; ++j2)
        #pragma unroll
        for (int r = 0; r < 8; ++r) o[j2].e[r] = 0.f;
    #pragma unroll
    for (int kk = 0; kk < 2; ++kk) {
        H16 ap;
        const _Float16* lp = &lds[wave][mi][32 * kk + 16 * khf];
        ap.h2[0] = *(const h8*)(lp);
        ap.h2[1] = *(const h8*)(lp + 8);
        #pragma unroll
        for (int j2 = 0; j2 < 2; ++j2) {
            int d = 16 * j2 + mi;
            H16 bf;
            #pragma unroll
            for (int i = 0; i < 16; ++i)
                bf.e[i] = vtile[32 * kk + 16 * khf + i][d];
            o[j2].v = __builtin_amdgcn_wmma_f32_16x16x32_f16(
                false, ap.v, false, bf.v, (short)0, o[j2].v, false, false);
        }
    }
    #pragma unroll
    for (int j2 = 0; j2 < 2; ++j2) {
        int d = 16 * j2 + mi;
        #pragma unroll
        for (int r = 0; r < 8; ++r) {
            int wrow = wBase + r + 8 * khf;
            vvT[((((size_t)b * 64 + wrow) * 8 + n) * 64 + h) * 32 + d] = (_Float16)o[j2].e[r];
        }
    }
}

// ---------------------------------------------------------------------------
// Attention along H: one block per (b, w, head).  vv tile (64x32, contiguous
// in vvT) async-copied to LDS; result stored to Uh in (b,h,w,c) layout.
// ---------------------------------------------------------------------------
__global__ __launch_bounds__(128)
void attn_h_kernel(const _Float16* __restrict__ Qh, const _Float16* __restrict__ Kh,
                   const _Float16* __restrict__ vvT, const float* __restrict__ mask_h,
                   _Float16* __restrict__ Uh)
{
    __shared__ _Float16 lds[4][16][64];
    __shared__ _Float16 gtile[64][32];            // [g][d]
    const int lane = threadIdx.x & 31;
    const int wave = threadIdx.x >> 5;
    const int mi   = lane & 15;
    const int khf  = lane >> 4;
    const int bid  = blockIdx.x;                 // (b*64+w)*8+n
    const int n = bid & 7, w = (bid >> 3) & 63, b = bid >> 9;
    const int hBase = wave * 16;
    const size_t base0 = (size_t)b * 1048576 + (size_t)w * 256 + (size_t)n * 32;
    const size_t vbase = (((size_t)b * 64 + w) * 8 + n) * 2048;   // [g][d] tile

    // async stage of the 4KB vv tile (fully contiguous in vvT)
    {
        int t = threadIdx.x;
        const _Float16* g = vvT + vbase + (size_t)t * 16;
        unsigned lo = lds_offset(&gtile[0][0]) + t * 32;
        async_ld_b128(lo, g);
        async_ld_b128(lo + 16, g + 8);
    }

    H16 a;
    {
        const _Float16* p = Qh + base0 + (size_t)(hBase + mi) * 16384 + 16 * khf;
        a.h2[0] = *(const h8*)(p);
        a.h2[1] = *(const h8*)(p + 8);
    }
    F8 acc[4];
    #pragma unroll
    for (int j = 0; j < 4; ++j) {
        int g = 16 * j + mi;
        #pragma unroll
        for (int r = 0; r < 8; ++r)
            acc[j].e[r] = mask_h[((size_t)n * 64 + (hBase + r + 8 * khf)) * 64 + g];
        H16 bf;
        const _Float16* kp = Kh + base0 + (size_t)g * 16384 + 16 * khf;
        bf.h2[0] = *(const h8*)(kp);
        bf.h2[1] = *(const h8*)(kp + 8);
        acc[j].v = __builtin_amdgcn_wmma_f32_16x16x32_f16(
            false, a.v, false, bf.v, (short)0, acc[j].v, false, false);
    }
    #pragma unroll
    for (int r = 0; r < 8; ++r) {
        float mx = fmaxf(fmaxf(acc[0].e[r], acc[1].e[r]), fmaxf(acc[2].e[r], acc[3].e[r]));
        #pragma unroll
        for (int off = 1; off < 16; off <<= 1) mx = fmaxf(mx, __shfl_xor(mx, off, 32));
        float sum = 0.f;
        #pragma unroll
        for (int j = 0; j < 4; ++j) { acc[j].e[r] = __expf(acc[j].e[r] - mx); sum += acc[j].e[r]; }
        #pragma unroll
        for (int off = 1; off < 16; off <<= 1) sum += __shfl_xor(sum, off, 32);
        float inv = 1.f / sum;
        #pragma unroll
        for (int j = 0; j < 4; ++j) acc[j].e[r] *= inv;
    }
    #pragma unroll
    for (int j = 0; j < 4; ++j)
        #pragma unroll
        for (int r = 0; r < 8; ++r)
            lds[wave][r + 8 * khf][16 * j + mi] = (_Float16)acc[j].e[r];

    wait_async0();
    __syncthreads();

    F8 o[2];
    #pragma unroll
    for (int j2 = 0; j2 < 2; ++j2)
        #pragma unroll
        for (int r = 0; r < 8; ++r) o[j2].e[r] = 0.f;
    #pragma unroll
    for (int kk = 0; kk < 2; ++kk) {
        H16 ap;
        const _Float16* lp = &lds[wave][mi][32 * kk + 16 * khf];
        ap.h2[0] = *(const h8*)(lp);
        ap.h2[1] = *(const h8*)(lp + 8);
        #pragma unroll
        for (int j2 = 0; j2 < 2; ++j2) {
            int d = 16 * j2 + mi;
            H16 bf;
            #pragma unroll
            for (int i = 0; i < 16; ++i)
                bf.e[i] = gtile[32 * kk + 16 * khf + i][d];
            o[j2].v = __builtin_amdgcn_wmma_f32_16x16x32_f16(
                false, ap.v, false, bf.v, (short)0, o[j2].v, false, false);
        }
    }
    #pragma unroll
    for (int j2 = 0; j2 < 2; ++j2) {
        int d = 16 * j2 + mi;
        #pragma unroll
        for (int r = 0; r < 8; ++r) {
            int hrow = hBase + r + 8 * khf;
            Uh[base0 + (size_t)hrow * 16384 + d] = (_Float16)o[j2].e[r];
        }
    }
}

// ---------------------------------------------------------------------------
// Depthwise 5x5 conv (lepe) on V, added in-place into Uh.
// ---------------------------------------------------------------------------
__global__ void lepe_kernel(const _Float16* __restrict__ Vh, const float* __restrict__ cw,
                            const float* __restrict__ cb, _Float16* __restrict__ Uh)
{
    size_t idx = (size_t)blockIdx.x * 256 + threadIdx.x;   // [0, B*H*W*C)
    int c = (int)(idx & 255);
    size_t row = idx >> 8;
    int w = (int)(row & 63);
    int h = (int)((row >> 6) & 63);
    int b = (int)(row >> 12);
    float acc = cb[c];
    #pragma unroll
    for (int kh2 = 0; kh2 < 5; ++kh2) {
        int hi = h + kh2 - 2;
        if (hi < 0 || hi >= 64) continue;
        #pragma unroll
        for (int kw2 = 0; kw2 < 5; ++kw2) {
            int wi = w + kw2 - 2;
            if (wi < 0 || wi >= 64) continue;
            acc += (float)Vh[(((size_t)b * 64 + hi) * 64 + wi) * 256 + c]
                 * cw[(kh2 * 5 + kw2) * 256 + c];
        }
    }
    Uh[idx] = (_Float16)((float)Uh[idx] + acc);
}

// ---------------------------------------------------------------------------
// Output projection: Uh (f16) @ Wo + bo -> d_out (f32).
// ---------------------------------------------------------------------------
__global__ __launch_bounds__(128)
void out_proj_kernel(const _Float16* __restrict__ Uh, const _Float16* __restrict__ Wo_pk,
                     const float* __restrict__ bo, float* __restrict__ out)
{
    const int lane = threadIdx.x & 31;
    const int wave = threadIdx.x >> 5;
    const int mi   = lane & 15;
    const int khf  = lane >> 4;
    const int rowBase = blockIdx.x * 64 + wave * 16;
    const int colBase = blockIdx.y * 64;

    F8 acc[4];
    #pragma unroll
    for (int j = 0; j < 4; ++j) {
        float bval = bo[colBase + 16 * j + mi];
        #pragma unroll
        for (int r = 0; r < 8; ++r) acc[j].e[r] = bval;
    }
    const _Float16* urow = Uh + (size_t)(rowBase + mi) * Cx;
    #pragma unroll
    for (int s = 0; s < 8; ++s) {
        H16 a;
        const _Float16* p0 = urow + 32 * s + 16 * khf;
        a.h2[0] = *(const h8*)(p0);
        a.h2[1] = *(const h8*)(p0 + 8);
        #pragma unroll
        for (int j = 0; j < 4; ++j) {
            const _Float16* bp = Wo_pk + (size_t)(2 * s + khf) * 4096
                                        + (size_t)(colBase + 16 * j + mi) * 16;
            H16 bf;
            bf.h2[0] = *(const h8*)(bp);
            bf.h2[1] = *(const h8*)(bp + 8);
            acc[j].v = __builtin_amdgcn_wmma_f32_16x16x32_f16(
                false, a.v, false, bf.v, (short)0, acc[j].v, false, false);
        }
    }
    #pragma unroll
    for (int j = 0; j < 4; ++j) {
        int col = colBase + 16 * j + mi;
        #pragma unroll
        for (int r = 0; r < 8; ++r)
            out[(size_t)(rowBase + r + 8 * khf) * Cx + col] = acc[j].e[r];
    }
}

// ---------------------------------------------------------------------------
extern "C" void kernel_launch(void* const* d_in, const int* in_sizes, int n_in,
                              void* d_out, int out_size, void* d_ws, size_t ws_size,
                              hipStream_t stream) {
    const float* x      = (const float*)d_in[0];
    const float* sinp   = (const float*)d_in[1];
    const float* cosp   = (const float*)d_in[2];
    const float* mask_h = (const float*)d_in[3];
    const float* mask_w = (const float*)d_in[4];
    const float* Wq     = (const float*)d_in[5];
    const float* bq     = (const float*)d_in[6];
    const float* Wk     = (const float*)d_in[7];
    const float* bk     = (const float*)d_in[8];
    const float* Wv     = (const float*)d_in[9];
    const float* bv     = (const float*)d_in[10];
    const float* conv_w = (const float*)d_in[11];
    const float* conv_b = (const float*)d_in[12];
    const float* Wo     = (const float*)d_in[13];
    const float* bo     = (const float*)d_in[14];
    float* out = (float*)d_out;

    // Workspace layout (all f16): 5 x (65536*256) tensors + 4 packed weights.
    char* ws = (char*)d_ws;
    const size_t T = (size_t)NROW * Cx * sizeof(_Float16);   // 32 MiB each
    _Float16* Qh    = (_Float16*)(ws + 0 * T);
    _Float16* Kh    = (_Float16*)(ws + 1 * T);
    _Float16* Vh    = (_Float16*)(ws + 2 * T);
    _Float16* vvT   = (_Float16*)(ws + 3 * T);
    _Float16* Uh    = (_Float16*)(ws + 4 * T);
    _Float16* Wpk   = (_Float16*)(ws + 5 * T);               // 4 x 65536 halfs
    _Float16* Wq_pk = Wpk + 0 * 65536;
    _Float16* Wk_pk = Wpk + 1 * 65536;
    _Float16* Wv_pk = Wpk + 2 * 65536;
    _Float16* Wo_pk = Wpk + 3 * 65536;

    // 1. pack weights into WMMA B layout
    pack_w_kernel<<<1024, 256, 0, stream>>>(Wq, Wk, Wv, Wo, Wpk);

    // 2. fused Q/K/V projection (K pre-scaled)
    proj_qkv_kernel<<<dim3(NROW / 64, Cx / 64), 128, 0, stream>>>(
        x, Wq_pk, bq, Wk_pk, bk, Wv_pk, bv, Qh, Kh, Vh);

    // 3. rotary (theta_shift) in-place on Q and K
    theta_kernel<<<(NROW * Cx / 2) / 256, 256, 0, stream>>>(Qh, Kh, sinp, cosp);

    // 4. attention along W -> vvT[b][w][n][h][d]
    attn_w_kernel<<<Bx * Hx * NHx, 128, 0, stream>>>(Qh, Kh, Vh, mask_w, vvT);

    // 5. attention along H -> Uh[b][h][w][c]
    attn_h_kernel<<<Bx * Wx * NHx, 128, 0, stream>>>(Qh, Kh, vvT, mask_h, Uh);

    // 6. lepe (depthwise 5x5 conv of V) added into Uh
    lepe_kernel<<<(NROW * Cx) / 256, 256, 0, stream>>>(Vh, conv_w, conv_b, Uh);

    // 7. output projection -> fp32 d_out
    out_proj_kernel<<<dim3(NROW / 64, Cx / 64), 128, 0, stream>>>(Uh, Wo_pk, bo, out);
}
